// RPN_34256659153117
// MI455X (gfx1250) — matrix operand order repaired and checked by reading, not compiled, not run on hardware
//
#include <hip/hip_runtime.h>
#include <hip/hip_bf16.h>
#include <math.h>

// ---------------------------------------------------------------------------
// MI455X (gfx1250) RPN: bf16 WMMA implicit-GEMM conv3x3 (zero-halo padded
// input, branch-free inner loop, async global->LDS A-tile staging) +
// WMMA 1x1 heads + refine + rank sort + greedy NMS.
// Wave32, V_WMMA_F32_16X16X32_BF16 (fp32 accum), ASYNCcnt-tracked LDS DMA.
// ---------------------------------------------------------------------------

typedef __attribute__((ext_vector_type(8)))  __bf16 v8bf;
typedef __attribute__((ext_vector_type(16))) __bf16 v16bf;
typedef __attribute__((ext_vector_type(8)))  float  v8f;

#define C_IN   1024
#define C_OUT  512
#define HW     64
#define NPIX   4096      // 64*64
#define K3     9216      // 1024*9
#define NANCH  36864     // 9*4096

// padded input plane: y,x in [-1,64] mapped to [0,65]/[0,71]; rest zero
#define PH     66
#define PW     72
#define PPLANE (PH * PW) // 4752

// LDS tile strides (elements): 80B rows -> 16B aligned, 20-bank stride
#define AST    40
#define BST    40

// gfx1250 async global->LDS copy path (ASYNCcnt).
// Probe result: builtin exists; arg0 type is int __attribute__((vector_size(16)))
// in AS1 (printed as "__device__"), non-const.
#if __has_builtin(__builtin_amdgcn_global_load_async_to_lds_b128) && \
    __has_builtin(__builtin_amdgcn_s_wait_asynccnt)
#define USE_ASYNC_LDS 1
typedef int v4i_vs __attribute__((vector_size(16)));
typedef __attribute__((address_space(1))) v4i_vs glb_v4i;
typedef __attribute__((address_space(3))) v4i_vs lds_v4i;
#define ASYNC_B128(gsrc, ldst, off) \
    __builtin_amdgcn_global_load_async_to_lds_b128( \
        (glb_v4i*)(v4i_vs*)(gsrc), (lds_v4i*)(v4i_vs*)(ldst), (off), 0)
#else
#define USE_ASYNC_LDS 0
#endif

__device__ __forceinline__ __bf16 f2bf(float f) {
    unsigned u = __float_as_uint(f);
    unsigned r = (u + 0x7FFFu + ((u >> 16) & 1u)) >> 16;
    unsigned short s = (unsigned short)r;
    return __builtin_bit_cast(__bf16, s);
}

// ---------------------------- conversion kernels ---------------------------

// fused fp32->bf16 convert + zero halo pad: xpad[1024][66][72]
__global__ void k_pad_x(const float* __restrict__ x, __bf16* __restrict__ xpad, int n) {
    int i = blockIdx.x * blockDim.x + threadIdx.x;
    if (i >= n) return;
    int ic  = i / PPLANE;
    int rem = i - ic * PPLANE;
    int yy  = rem / PW;       // 0..65
    int xx  = rem - yy * PW;  // 0..71
    int y = yy - 1, xp = xx - 1;
    float v = 0.f;
    if (y >= 0 && y < HW && xp >= 0 && xp < HW)
        v = x[(size_t)ic * NPIX + y * HW + xp];
    xpad[i] = f2bf(v);
}

// conv_w (OIHW: [512][1024][3][3]) -> tap-major [512][9][1024] bf16
__global__ void k_cvt_w(const float* __restrict__ w, __bf16* __restrict__ wb) {
    int i = blockIdx.x * blockDim.x + threadIdx.x;
    if (i < C_OUT * K3) {
        int oc  = i / K3;
        int rem = i - oc * K3;
        int tap = rem >> 10;
        int ic  = rem & 1023;
        wb[i] = f2bf(w[(size_t)oc * K3 + ic * 9 + tap]);
    }
}

// pack cls_w (18x512) + bbox_w (36x512) into 64x512 bf16 (rows 54..63 zero)
__global__ void k_cvt_heads(const float* __restrict__ cls_w,
                            const float* __restrict__ bbox_w,
                            __bf16* __restrict__ wc) {
    int i = blockIdx.x * blockDim.x + threadIdx.x;
    if (i < 64 * 512) {
        int row = i >> 9, k = i & 511;
        float v = 0.f;
        if (row < 18)      v = cls_w[row * 512 + k];
        else if (row < 54) v = bbox_w[(row - 18) * 512 + k];
        wc[i] = f2bf(v);
    }
}

// --------------------- conv3x3 implicit GEMM (WMMA bf16) -------------------
// C[512][4096] = sum_tap sum_ic W[oc][tap*1024+ic] * Xpad_shift[ic][px]
// Grid: (64 n-tiles = one image row each, 4 m-tiles). Block 256 = 8 waves.
// WG tile 128x64, wave tile 32x32 (2x2 WMMA), BK=32.

__global__ __launch_bounds__(256)
void k_conv3x3_wmma(const __bf16* __restrict__ xpad,  // [1024][66][72]
                    const __bf16* __restrict__ wb,    // [512][9][1024]
                    const float*  __restrict__ conv_b,
                    __bf16* __restrict__ hb) {        // [512][4096]
    __shared__ __attribute__((aligned(16))) __bf16 As[128 * AST];
    __shared__ __attribute__((aligned(16))) __bf16 Bs[64 * BST];

    const int tid  = threadIdx.x;
    const int lane = tid & 31;
    const int wave = tid >> 5;
    const int wm   = wave & 3;        // 0..3
    const int wn   = wave >> 2;       // 0..1
    const int m0   = blockIdx.y * 128;
    const int yrow = blockIdx.x;      // n-tile == one image row
    const int n0   = yrow * 64;

    v8f acc[2][2] = {};

    const int a_row = tid >> 1;            // 0..127
    const int a_seg = (tid & 1) * 16;      // 0 / 16
    const int b_k   = tid >> 3;            // 0..31
    const int b_n   = (tid & 7) * 8;       // 0..56
    const int h16   = (lane >> 4) & 1;
    const int l15   = lane & 15;

    for (int tap = 0; tap < 9; ++tap) {
        const int dy = tap / 3 - 1;
        const int dx = tap % 3 - 1;
        const int prow = (yrow + dy + 1) * PW + (dx + 1); // halo keeps in-range
        for (int ic0 = 0; ic0 < C_IN; ic0 += 32) {
            // ---- A tile (weights) 128x32 ----
            {
                const __bf16* src = wb + (size_t)(m0 + a_row) * K3 + tap * 1024 + ic0 + a_seg;
                __bf16* dst = As + a_row * AST + a_seg;
#if USE_ASYNC_LDS
                // global -> LDS DMA, tracked by ASYNCcnt (no VGPR round trip)
                ASYNC_B128(src, dst, 0);
                ASYNC_B128(src, dst, 16);
#else
                *(v8bf*)(dst)     = *(const v8bf*)(src);
                *(v8bf*)(dst + 8) = *(const v8bf*)(src + 8);
#endif
                __builtin_prefetch(src + 32, 0, 1);   // gfx1250 global_prefetch_b8
            }
            // ---- B tile: branch-free 16B load from halo, transpose to [n][k] ----
            {
                const __bf16* src = xpad + (size_t)(ic0 + b_k) * PPLANE + prow + b_n;
                v8bf vals;
                __builtin_memcpy(&vals, src, sizeof(vals));  // may be 2B-aligned
                #pragma unroll
                for (int e = 0; e < 8; ++e)
                    Bs[(b_n + e) * BST + b_k] = vals[e];
            }
#if USE_ASYNC_LDS
            __builtin_amdgcn_s_wait_asynccnt(0);
#endif
            __syncthreads();

            // ---- WMMA: 2x2 tiles per wave ----
            #pragma unroll
            for (int sm = 0; sm < 2; ++sm) {
                const int mr = wm * 32 + sm * 16 + l15;
                union { v16bf v; v8bf h[2]; } A;
                // ISA 16-bit A 16x32 layout: K = e + 8*(lane>=16) + 8*(e>=8)
                A.h[0] = *(const v8bf*)(As + mr * AST + 8 * h16);
                A.h[1] = *(const v8bf*)(As + mr * AST + 16 + 8 * h16);
                #pragma unroll
                for (int sn = 0; sn < 2; ++sn) {
                    const int nr = wn * 32 + sn * 16 + l15;
                    union { v16bf v; v8bf h[2]; } B;
                    // B (K x N): lane -> N, elements -> K = e + 8*(lane>=16) + 8*(e>=8)
                    B.h[0] = *(const v8bf*)(Bs + nr * BST + 8 * h16);
                    B.h[1] = *(const v8bf*)(Bs + nr * BST + 16 + 8 * h16);
                    acc[sm][sn] = __builtin_amdgcn_wmma_f32_16x16x32_bf16(
                        false, A.v, false, B.v, (short)0, acc[sm][sn], false, false);
                }
            }
            __syncthreads();
        }
    }

    // epilogue: bias + ReLU -> bf16 h
    #pragma unroll
    for (int sm = 0; sm < 2; ++sm) {
        #pragma unroll
        for (int sn = 0; sn < 2; ++sn) {
            const int ng = n0 + wn * 32 + sn * 16 + l15;
            #pragma unroll
            for (int r = 0; r < 8; ++r) {
                const int mg = m0 + wm * 32 + sm * 16 + r + 8 * h16;
                float v = acc[sm][sn][r] + conv_b[mg];
                v = v > 0.f ? v : 0.f;
                hb[(size_t)mg * NPIX + ng] = f2bf(v);
            }
        }
    }
}

// -------------------- 1x1 heads GEMM (WMMA bf16, M=64) ---------------------

__global__ __launch_bounds__(128)
void k_heads_wmma(const __bf16* __restrict__ hb,    // [512][4096]
                  const __bf16* __restrict__ wc,    // [64][512]
                  const float* __restrict__ cls_b,
                  const float* __restrict__ bbox_b,
                  float* __restrict__ probs_out,    // [18][4096] (into d_out)
                  float* __restrict__ bboxp) {      // [36][4096] (ws)
    __shared__ __attribute__((aligned(16))) __bf16 As[64 * AST];
    __shared__ __attribute__((aligned(16))) __bf16 Bs[64 * BST];

    const int tid  = threadIdx.x;
    const int lane = tid & 31;
    const int wave = tid >> 5;            // 0..3, one 16-row strip each
    const int n0   = blockIdx.x * 64;
    const int h16  = (lane >> 4) & 1;
    const int l15  = lane & 15;

    v8f acc[4] = {};

    const int a_row = tid >> 1;           // 0..63
    const int a_seg = (tid & 1) * 16;
    const int b_k   = tid >> 2;           // 0..31
    const int b_n   = (tid & 3) * 16;

    for (int k0 = 0; k0 < 512; k0 += 32) {
        {
            const __bf16* src = wc + a_row * 512 + k0 + a_seg;
            __bf16* dst = As + a_row * AST + a_seg;
#if USE_ASYNC_LDS
            ASYNC_B128(src, dst, 0);
            ASYNC_B128(src, dst, 16);
#else
            *(v8bf*)(dst)     = *(const v8bf*)(src);
            *(v8bf*)(dst + 8) = *(const v8bf*)(src + 8);
#endif
        }
        {
            const __bf16* src = hb + (size_t)(k0 + b_k) * NPIX + n0 + b_n;
            v8bf lo = *(const v8bf*)(src);
            v8bf hi = *(const v8bf*)(src + 8);
            #pragma unroll
            for (int e = 0; e < 8; ++e) {
                Bs[(b_n + e)     * BST + b_k] = lo[e];
                Bs[(b_n + e + 8) * BST + b_k] = hi[e];
            }
        }
#if USE_ASYNC_LDS
        __builtin_amdgcn_s_wait_asynccnt(0);
#endif
        __syncthreads();

        const int mr = wave * 16 + l15;
        union { v16bf v; v8bf h[2]; } A;
        A.h[0] = *(const v8bf*)(As + mr * AST + 8 * h16);
        A.h[1] = *(const v8bf*)(As + mr * AST + 16 + 8 * h16);
        #pragma unroll
        for (int sn = 0; sn < 4; ++sn) {
            const int nr = sn * 16 + l15;
            union { v16bf v; v8bf h[2]; } B;
            B.h[0] = *(const v8bf*)(Bs + nr * BST + 8 * h16);
            B.h[1] = *(const v8bf*)(Bs + nr * BST + 16 + 8 * h16);
            acc[sn] = __builtin_amdgcn_wmma_f32_16x16x32_bf16(
                false, A.v, false, B.v, (short)0, acc[sn], false, false);
        }
        __syncthreads();
    }

    #pragma unroll
    for (int sn = 0; sn < 4; ++sn) {
        const int ng = n0 + sn * 16 + l15;
        #pragma unroll
        for (int r = 0; r < 8; ++r) {
            const int ch = wave * 16 + r + 8 * h16;
            float v = acc[sn][r];
            if (ch < 18) {
                v += cls_b[ch];
                probs_out[ch * NPIX + ng] = 1.f / (1.f + expf(-v));
            } else if (ch < 54) {
                int c = ch - 18;
                bboxp[c * NPIX + ng] = v + bbox_b[c];
            }
        }
    }
}

// --------------------------- anchors / refine ------------------------------

__device__ __forceinline__ void anchor_box(int a, int px, float* o) {
    const float S[3] = {48.f, 64.f, 80.f};
    const float R[3] = {0.6f, 1.0f, 1.5f};
    int si = a / 3, ri = a % 3;
    float sr = sqrtf(R[ri]);
    float w = S[si] * sr, h = S[si] / sr;
    float cx = (float)((px & 63) * 8 + 4);
    float cy = (float)((px >> 6) * 8 + 4);
    o[0] = cx - 0.5f * w; o[1] = cy - 0.5f * h;
    o[2] = cx + 0.5f * w; o[3] = cy + 0.5f * h;
}

__global__ void k_refine(const float* __restrict__ probs,   // [18][4096]
                         const float* __restrict__ bboxp,   // [36][4096]
                         float* __restrict__ refined,       // [N][4]
                         float* __restrict__ keys,          // [N]
                         int*   __restrict__ validb) {      // [N]
    int i = blockIdx.x * blockDim.x + threadIdx.x;
    if (i >= NANCH) return;
    // scores/anchors anchor-major; deltas pixel-major (reference semantics)
    int a = i >> 12, px = i & 4095;
    float s = probs[(2 * a + 1) * NPIX + px];
    int valid = s > 0.5f;
    validb[i] = valid;
    keys[i]   = valid ? s : -INFINITY;

    float an[4];
    anchor_box(a, px, an);
    int pxd = i / 9, ad = i - pxd * 9;
    float d0 = bboxp[(ad * 4 + 0) * NPIX + pxd];
    float d1 = bboxp[(ad * 4 + 1) * NPIX + pxd];
    float d2 = bboxp[(ad * 4 + 2) * NPIX + pxd];
    float d3 = bboxp[(ad * 4 + 3) * NPIX + pxd];
    float w  = an[2] - an[0], h = an[3] - an[1];
    float cx = an[0] + 0.5f * w, cy = an[1] + 0.5f * h;
    float pcx = d0 * w + cx, pcy = d1 * h + cy;
    float pw  = expf(d2) * w, ph = expf(d3) * h;
    refined[i * 4 + 0] = pcx - 0.5f * pw;
    refined[i * 4 + 1] = pcy - 0.5f * ph;
    refined[i * 4 + 2] = pcx + 0.5f * pw;
    refined[i * 4 + 3] = pcy + 0.5f * ph;
}

// stable descending argsort by rank counting (N = 144*256 exactly)
__global__ __launch_bounds__(256)
void k_rank(const float* __restrict__ keys, int* __restrict__ order) {
    __shared__ float sk[256];
    int i = blockIdx.x * 256 + threadIdx.x;
    float ki = keys[i];
    int rank = 0;
    for (int base = 0; base < NANCH; base += 256) {
        sk[threadIdx.x] = keys[base + threadIdx.x];
        __syncthreads();
        #pragma unroll 8
        for (int t = 0; t < 256; ++t) {
            int j = base + t;
            float kj = sk[t];
            rank += (kj > ki) || (kj == ki && j < i);
        }
        __syncthreads();
    }
    order[rank] = i;
}

// sequential greedy NMS, one workgroup (matches reference fori_loop)
__global__ __launch_bounds__(1024)
void k_nms(const float* __restrict__ refined,
           const int*   __restrict__ order,
           const int*   __restrict__ validb,
           unsigned*    __restrict__ suppressed,
           float* __restrict__ prop_out,    // [N][4]
           float* __restrict__ keep_out) {  // [N]
    for (int j = threadIdx.x; j < NANCH; j += 1024) suppressed[j] = 0u;
    __syncthreads();

    __shared__ float sb[4];
    __shared__ int   stake;

    for (int i = 0; i < NANCH; ++i) {
        if (threadIdx.x == 0) {
            int oi = order[i];
            int take = validb[oi] && !suppressed[i];
            stake = take;
            const float* bi = refined + (size_t)oi * 4;
            sb[0] = bi[0]; sb[1] = bi[1]; sb[2] = bi[2]; sb[3] = bi[3];
            keep_out[i] = take ? 1.f : 0.f;
            float an[4] = {0.f, 0.f, 0.f, 0.f};
            if (take) anchor_box(oi >> 12, oi & 4095, an);
            prop_out[i * 4 + 0] = an[0];
            prop_out[i * 4 + 1] = an[1];
            prop_out[i * 4 + 2] = an[2];
            prop_out[i * 4 + 3] = an[3];
        }
        __syncthreads();
        if (stake) {
            const float x1 = sb[0], y1 = sb[1], x2 = sb[2], y2 = sb[3];
            const float ai = (x2 - x1) * (y2 - y1);
            for (int j = i + 1 + threadIdx.x; j < NANCH; j += 1024) {
                if (suppressed[j]) continue;
                const float* bj = refined + (size_t)order[j] * 4;
                float xx1 = fmaxf(x1, bj[0]), yy1 = fmaxf(y1, bj[1]);
                float xx2 = fminf(x2, bj[2]), yy2 = fminf(y2, bj[3]);
                float inter = fmaxf(xx2 - xx1, 0.f) * fmaxf(yy2 - yy1, 0.f);
                float aj = (bj[2] - bj[0]) * (bj[3] - bj[1]);
                float iou = inter / (aj + ai - inter);
                if (iou > 0.3f) suppressed[j] = 1u;
            }
        }
        __syncthreads();
    }
}

// ------------------------------- launcher ----------------------------------

extern "C" void kernel_launch(void* const* d_in, const int* in_sizes, int n_in,
                              void* d_out, int out_size, void* d_ws, size_t ws_size,
                              hipStream_t stream) {
    const float* x      = (const float*)d_in[0];
    const float* conv_w = (const float*)d_in[1];
    const float* conv_b = (const float*)d_in[2];
    const float* cls_w  = (const float*)d_in[3];
    const float* cls_b  = (const float*)d_in[4];
    const float* bbox_w = (const float*)d_in[5];
    const float* bbox_b = (const float*)d_in[6];
    (void)in_sizes; (void)n_in; (void)out_size; (void)ws_size;

    char* ws = (char*)d_ws;
    size_t off = 0;
    auto walloc = [&](size_t bytes) -> void* {
        off = (off + 255) & ~(size_t)255;
        void* p = ws + off;
        off += bytes;
        return p;
    };
    __bf16*   xpad  = (__bf16*)  walloc((size_t)C_IN * PPLANE * 2);  // 9.3 MB
    __bf16*   wb    = (__bf16*)  walloc((size_t)C_OUT * K3 * 2);     // 9 MB
    __bf16*   hb    = (__bf16*)  walloc((size_t)C_OUT * NPIX * 2);   // 4 MB
    __bf16*   wc    = (__bf16*)  walloc((size_t)64 * 512 * 2);
    float*    bboxp = (float*)   walloc((size_t)36 * NPIX * 4);
    float*    refnd = (float*)   walloc((size_t)NANCH * 4 * 4);
    float*    keys  = (float*)   walloc((size_t)NANCH * 4);
    int*      valid = (int*)     walloc((size_t)NANCH * 4);
    int*      order = (int*)     walloc((size_t)NANCH * 4);
    unsigned* supp  = (unsigned*)walloc((size_t)NANCH * 4);

    float* out      = (float*)d_out;
    float* prop_out = out;                       // 36864*4
    float* keep_out = out + NANCH * 4;           // 36864
    float* probs_out= out + NANCH * 4 + NANCH;   // 18*4096

    // 1) conversions: pad+convert x, repack weights
    {
        int n = C_IN * PPLANE;
        k_pad_x<<<dim3((n + 255) / 256), dim3(256), 0, stream>>>(x, xpad, n);
    }
    {
        int n = C_OUT * K3;
        k_cvt_w<<<dim3((n + 255) / 256), dim3(256), 0, stream>>>(conv_w, wb);
    }
    k_cvt_heads<<<dim3((64 * 512) / 256), dim3(256), 0, stream>>>(cls_w, bbox_w, wc);

    // 2) 3x3 conv + ReLU via WMMA implicit GEMM
    k_conv3x3_wmma<<<dim3(64, 4), dim3(256), 0, stream>>>(xpad, wb, conv_b, hb);

    // 3) fused 1x1 heads via WMMA (+bias, sigmoid)
    k_heads_wmma<<<dim3(64), dim3(128), 0, stream>>>(hb, wc, cls_b, bbox_b, probs_out, bboxp);

    // 4) box refinement + validity + sort keys
    k_refine<<<dim3(NANCH / 256), dim3(256), 0, stream>>>(probs_out, bboxp, refnd, keys, valid);

    // 5) stable descending argsort (rank counting)
    k_rank<<<dim3(NANCH / 256), dim3(256), 0, stream>>>(keys, order);

    // 6) greedy NMS + proposals/keep outputs
    k_nms<<<dim3(1), dim3(1024), 0, stream>>>(refnd, order, valid, supp, prop_out, keep_out);
}
